// GeneticDiffusionModuleBlock_1726576856762
// MI455X (gfx1250) — compile-verified, matrix operand order checked by reading, not run on hardware
//
#include <hip/hip_runtime.h>
#include <hip/hip_bf16.h>

// Only the last diffusion step survives the reference's scan (carry is
// overwritten each iteration), so this is a single fused noisy-MLP pass:
//   y = relu((x + N(0,1)*scale[63]) @ W1 + b1) @ W2 + b2
// via v_wmma_f32_16x16x32_f16 (f16 in, f32 accumulate).
//
// Strategy (MI455X): one-time prep kernel converts W1/W2 to f16 and
// pre-transposes them to [N][K] in d_ws (512 KB, L2-resident). The fused
// kernel then loads B fragments straight from L2 with global_load_b128 —
// no weight staging, no per-slab barriers. A fragments come from LDS
// (noisy x / relu hidden, both f16).

typedef __attribute__((ext_vector_type(16))) _Float16 v16h;
typedef __attribute__((ext_vector_type(8)))  _Float16 v8h;
typedef __attribute__((ext_vector_type(8)))  float    v8f;

#define C_IN      256
#define C_HID     512
#define M_TOT     8192      // 2*64*64 tokens
#define M_BLK     64        // token strip per workgroup
#define K_SLAB    32        // K per f16 WMMA
#define NSTEPS    64

__device__ __forceinline__ unsigned pcg_hash(unsigned v) {
    v = v * 747796405u + 2891336453u;
    unsigned w = ((v >> ((v >> 28u) + 4u)) ^ v) * 277803737u;
    return (w >> 22u) ^ w;
}

// Box-Muller normal from hashed uniforms — structural stand-in for
// jax.random.normal(fold_in(key42, 63)); exact threefry bit-match impossible.
__device__ __forceinline__ float gauss_noise(unsigned idx) {
    unsigned h1 = pcg_hash(idx * 2u + 0x9E3779B9u);
    unsigned h2 = pcg_hash(idx * 2u + 1u);
    float u1 = ((float)h1 + 0.5f) * 2.3283064365386963e-10f; // (0,1)
    float u2 = ((float)h2 + 0.5f) * 2.3283064365386963e-10f;
    float r  = __fsqrt_rn(-2.0f * __logf(u1));
    return r * __cosf(6.2831853071795864f * u2);
}

__device__ __forceinline__ v16h cat8(v8h lo, v8h hi) {
    return __builtin_shufflevector(lo, hi, 0,1,2,3,4,5,6,7,8,9,10,11,12,13,14,15);
}

__device__ __forceinline__ v16h ldfrag(const _Float16* p) {
    // 16 halves per lane = two contiguous 16B chunks (K layout of 16-bit WMMA A/B)
    v8h lo = *reinterpret_cast<const v8h*>(p);
    v8h hi = *reinterpret_cast<const v8h*>(p + 16);
    return cat8(lo, hi);
}

// ---- one-time prep: W[K][N] f32 -> Wt[N][K] f16 (LDS-tiled transpose) -------
__global__ __launch_bounds__(256)
void cvt_transpose(const float* __restrict__ W, _Float16* __restrict__ Wt,
                   int K, int N) {
    __shared__ _Float16 t[32][33];
    const int ntile = N / 32;
    const int bx = blockIdx.x % ntile;   // col tile (n)
    const int by = blockIdx.x / ntile;   // row tile (k)
    const int tx = threadIdx.x & 31;
    const int ty = threadIdx.x >> 5;     // 0..7
#pragma unroll
    for (int r = 0; r < 32; r += 8)
        t[ty + r][tx] = (_Float16)W[(size_t)(by * 32 + ty + r) * N + bx * 32 + tx];
    __syncthreads();
#pragma unroll
    for (int r = 0; r < 32; r += 8)
        Wt[(size_t)(bx * 32 + ty + r) * K + by * 32 + tx] = t[tx][ty + r];
}

// ---- fused noisy MLP --------------------------------------------------------
__global__ __launch_bounds__(256)
void diffusion_mlp_fused(const float* __restrict__ x,
                         const float* __restrict__ noise_scale,
                         const _Float16* __restrict__ W1t,  // [C_HID][C_IN]
                         const float* __restrict__ b1,
                         const _Float16* __restrict__ W2t,  // [C_IN][C_HID]
                         const float* __restrict__ b2,
                         float* __restrict__ out) {
    extern __shared__ __align__(16) _Float16 smem[];
    _Float16* sA = smem;                  // [M_BLK][C_IN]   32 KB (noisy x, f16)
    _Float16* sH = sA + M_BLK * C_IN;     // [M_BLK][C_HID]  64 KB (relu hidden, f16)

    const int tid      = threadIdx.x;
    const int lane     = tid & 31;
    const int wave     = tid >> 5;        // 0..7
    const int laneM    = lane & 15;
    const int laneHalf = lane >> 4;       // 0/1
    const int kbase    = laneHalf * 8;    // per 16-bit WMMA A/B K striping
    const int m0       = blockIdx.x * M_BLK;

    const float scale = noise_scale[NSTEPS - 1];

    // stage noisy x strip -> f16 LDS, one ds_store_b128 per 8 elements
    for (int it = tid; it < M_BLK * C_IN / 8; it += 256) {
        const int s = it * 8;
        const float4 xa = *reinterpret_cast<const float4*>(x + (size_t)m0 * C_IN + s);
        const float4 xb = *reinterpret_cast<const float4*>(x + (size_t)m0 * C_IN + s + 4);
        const unsigned g = (unsigned)(m0 * C_IN + s);
        v8h h;
        h[0] = (_Float16)(xa.x + gauss_noise(g + 0u) * scale);
        h[1] = (_Float16)(xa.y + gauss_noise(g + 1u) * scale);
        h[2] = (_Float16)(xa.z + gauss_noise(g + 2u) * scale);
        h[3] = (_Float16)(xa.w + gauss_noise(g + 3u) * scale);
        h[4] = (_Float16)(xb.x + gauss_noise(g + 4u) * scale);
        h[5] = (_Float16)(xb.y + gauss_noise(g + 5u) * scale);
        h[6] = (_Float16)(xb.z + gauss_noise(g + 6u) * scale);
        h[7] = (_Float16)(xb.w + gauss_noise(g + 7u) * scale);
        *reinterpret_cast<v8h*>(sA + s) = h;
    }

    // ---- GEMM1: H[64,512] = relu(A @ W1 + b1); wave owns 64x64 --------------
    const int n0 = wave * 64;
    v8f acc[4][4];
#pragma unroll
    for (int nt = 0; nt < 4; ++nt) {
        const float bv = b1[n0 + nt * 16 + laneM];
#pragma unroll
        for (int mt = 0; mt < 4; ++mt)
#pragma unroll
            for (int i = 0; i < 8; ++i) acc[mt][nt][i] = bv;
    }

    __syncthreads();   // sA ready

    for (int ks = 0; ks < C_IN / K_SLAB; ++ks) {
        const int k0 = ks * K_SLAB;
        v16h afr[4], bfr[4];
#pragma unroll
        for (int mt = 0; mt < 4; ++mt)
            afr[mt] = ldfrag(sA + (mt * 16 + laneM) * C_IN + k0 + kbase);
#pragma unroll
        for (int nt = 0; nt < 4; ++nt)
            bfr[nt] = ldfrag(W1t + (size_t)(n0 + nt * 16 + laneM) * C_IN + k0 + kbase);
        if (ks + 1 < C_IN / K_SLAB)    // gfx1250 global_prefetch_b8 of next slab
            __builtin_prefetch(W1t + (size_t)(n0 + lane) * C_IN + k0 + K_SLAB, 0, 1);
#pragma unroll
        for (int mt = 0; mt < 4; ++mt)
#pragma unroll
            for (int nt = 0; nt < 4; ++nt)
                acc[mt][nt] = __builtin_amdgcn_wmma_f32_16x16x32_f16(
                    false, afr[mt], false, bfr[nt], (short)0, acc[mt][nt], false, false);
    }

    // relu -> f16 hidden strip in LDS
#pragma unroll
    for (int mt = 0; mt < 4; ++mt)
#pragma unroll
        for (int nt = 0; nt < 4; ++nt)
#pragma unroll
            for (int i = 0; i < 8; ++i) {
                const int row = mt * 16 + laneHalf * 8 + i;
                const int col = n0 + nt * 16 + laneM;
                const float v = acc[mt][nt][i];
                sH[row * C_HID + col] = (_Float16)(v > 0.0f ? v : 0.0f);
            }

    __syncthreads();   // sH ready

    // ---- GEMM2: Y[64,256] = H @ W2 + b2; wave owns 64x32 --------------------
    const int n0b = wave * 32;
    v8f acc2[4][2];
#pragma unroll
    for (int nt = 0; nt < 2; ++nt) {
        const float bv = b2[n0b + nt * 16 + laneM];
#pragma unroll
        for (int mt = 0; mt < 4; ++mt)
#pragma unroll
            for (int i = 0; i < 8; ++i) acc2[mt][nt][i] = bv;
    }

    for (int ks = 0; ks < C_HID / K_SLAB; ++ks) {
        const int k0 = ks * K_SLAB;
        v16h afr[4], bfr[2];
#pragma unroll
        for (int mt = 0; mt < 4; ++mt)
            afr[mt] = ldfrag(sH + (mt * 16 + laneM) * C_HID + k0 + kbase);
#pragma unroll
        for (int nt = 0; nt < 2; ++nt)
            bfr[nt] = ldfrag(W2t + (size_t)(n0b + nt * 16 + laneM) * C_HID + k0 + kbase);
        if (ks + 1 < C_HID / K_SLAB)
            __builtin_prefetch(W2t + (size_t)(n0b + lane) * C_HID + k0 + K_SLAB, 0, 1);
#pragma unroll
        for (int mt = 0; mt < 4; ++mt)
#pragma unroll
            for (int nt = 0; nt < 2; ++nt)
                acc2[mt][nt] = __builtin_amdgcn_wmma_f32_16x16x32_f16(
                    false, afr[mt], false, bfr[nt], (short)0, acc2[mt][nt], false, false);
    }

    // ---- store fp32 output --------------------------------------------------
#pragma unroll
    for (int mt = 0; mt < 4; ++mt)
#pragma unroll
        for (int nt = 0; nt < 2; ++nt)
#pragma unroll
            for (int i = 0; i < 8; ++i) {
                const int row = m0 + mt * 16 + laneHalf * 8 + i;
                const int col = n0b + nt * 16 + laneM;
                out[(size_t)row * C_IN + col] = acc2[mt][nt][i];
            }
}

extern "C" void kernel_launch(void* const* d_in, const int* in_sizes, int n_in,
                              void* d_out, int out_size, void* d_ws, size_t ws_size,
                              hipStream_t stream) {
    (void)in_sizes; (void)n_in; (void)out_size; (void)ws_size;
    const float* x  = (const float*)d_in[0];
    const float* ns = (const float*)d_in[1];
    const float* W1 = (const float*)d_in[2];
    const float* b1 = (const float*)d_in[3];
    const float* W2 = (const float*)d_in[4];
    const float* b2 = (const float*)d_in[5];
    float* out = (float*)d_out;

    _Float16* W1t = (_Float16*)d_ws;                       // [512][256] = 256 KB
    _Float16* W2t = W1t + (size_t)C_HID * C_IN;            // [256][512] = 256 KB

    // one-time weight convert+transpose (stream-ordered before the fused pass)
    hipLaunchKernelGGL(cvt_transpose, dim3((C_IN / 32) * (C_HID / 32)), dim3(256),
                       0, stream, W1, W1t, C_IN, C_HID);
    hipLaunchKernelGGL(cvt_transpose, dim3((C_HID / 32) * (C_IN / 32)), dim3(256),
                       0, stream, W2, W2t, C_HID, C_IN);

    const size_t shmem = (size_t)(M_BLK * C_IN + M_BLK * C_HID) * 2u;  // 96 KB
    hipLaunchKernelGGL(diffusion_mlp_fused,
                       dim3(M_TOT / M_BLK), dim3(256), shmem, stream,
                       x, ns, W1t, b1, W2t, b2, out);
}
